// Transformer_21251498180881
// MI455X (gfx1250) — matrix operand order repaired and checked by reading, not compiled, not run on hardware
//
#include <hip/hip_runtime.h>
#include <hip/hip_bf16.h>
#include <math.h>
#include <stdint.h>

#define B_  8
#define T_  512
#define D_  512
#define H_  8
#define FF_ 2048
#define L_  6
#define CS_ 16
#define DK_ 64

#define LDS_STRIDE 40   // bf16 elements per LDS tile row (80B: bank step 20 -> 16 conflict-free rows)

typedef __attribute__((ext_vector_type(16))) __bf16 bfrag;   // 16x16x32 bf16 A/B fragment (8 VGPRs)
typedef __attribute__((ext_vector_type(8)))  float  v8f;     // 16x16 f32 C/D tile (8 VGPRs)

union LoadU { uint4 u; __bf16 b[8]; };

// Load a WMMA fragment as two contiguous 16-byte chunks (p0 -> elems 0..7, p1 -> elems 8..15).
// A-matrix 16x32:  lane<16 holds K={0..7,16..23}, lane>=16 holds K={8..15,24..31} of row (lane&15).
// B-matrix 32x16:  lane<16 holds K={0..7,16..23}, lane>=16 holds K={8..15,24..31} of column (lane&15).
static __device__ inline bfrag load_frag(const __hip_bfloat16* p0, const __hip_bfloat16* p1) {
  LoadU lo, hi;
  lo.u = *reinterpret_cast<const uint4*>(p0);
  hi.u = *reinterpret_cast<const uint4*>(p1);
  bfrag f;
#pragma unroll
  for (int i = 0; i < 8; ++i) { f[i] = lo.b[i]; f[i + 8] = hi.b[i]; }
  return f;
}

static __device__ inline v8f wmma_bf16(bfrag a, bfrag b, v8f c) {
  // (neg_a, A, neg_b, B, c_mod, C, reuse_a, reuse_b)
  return __builtin_amdgcn_wmma_f32_16x16x32_bf16(false, a, false, b, (short)0, c, false, false);
}

// 16B async copy global -> LDS (ASYNCcnt-tracked, no VGPR data round trip).
static __device__ inline void async_load_b128(uint64_t gaddr, uint32_t lds_off) {
  asm volatile("global_load_async_to_lds_b128 %0, %1, off"
               :: "v"(lds_off), "v"(gaddr)
               : "memory");
}
static __device__ inline void wait_async0() {
  asm volatile("s_wait_asynccnt 0x0" ::: "memory");
}

// ---------------------------------------------------------------------------
// Elementwise helpers
// ---------------------------------------------------------------------------
__global__ void scale_kernel(const float* __restrict__ in, float* __restrict__ out,
                             int n, float s) {
  int i = blockIdx.x * blockDim.x + threadIdx.x;
  if (i < n) out[i] = in[i] * s;
}

__global__ void cvt_bf16_kernel(const float* __restrict__ in, __hip_bfloat16* __restrict__ out,
                                int n) {
  int i = blockIdx.x * blockDim.x + threadIdx.x;
  if (i < n) out[i] = __float2bfloat16(in[i]);
}

// qu = q + pbu, qv = q + pbv  (pbu/pbv broadcast over rows, period D)
__global__ void quqv_kernel(const float* __restrict__ q,
                            const float* __restrict__ pbu, const float* __restrict__ pbv,
                            __hip_bfloat16* __restrict__ quo, __hip_bfloat16* __restrict__ qvo,
                            int n, int D) {
  int i = blockIdx.x * blockDim.x + threadIdx.x;
  if (i < n) {
    int d = i % D;
    float v = q[i];
    quo[i] = __float2bfloat16(v + pbu[d]);
    qvo[i] = __float2bfloat16(v + pbv[d]);
  }
}

// ---------------------------------------------------------------------------
// Weight transpose + fp32 -> bf16:  W[K][N] -> Wt[N][K]   (batched over z)
// ---------------------------------------------------------------------------
__global__ void transpose_to_bf16(const float* __restrict__ W, __hip_bfloat16* __restrict__ Wt,
                                  int K, int N) {
  __shared__ float tile[32][33];
  const float* Wl = W + (size_t)blockIdx.z * K * N;
  __hip_bfloat16* Wtl = Wt + (size_t)blockIdx.z * K * N;
  int n0 = blockIdx.x * 32, k0 = blockIdx.y * 32;
  for (int i = threadIdx.y; i < 32; i += 8)
    tile[i][threadIdx.x] = Wl[(size_t)(k0 + i) * N + n0 + threadIdx.x];
  __syncthreads();
  for (int i = threadIdx.y; i < 32; i += 8)
    Wtl[(size_t)(n0 + i) * K + k0 + threadIdx.x] = __float2bfloat16(tile[threadIdx.x][i]);
}

// ---------------------------------------------------------------------------
// LayerNorm (one 256-thread block per row); optional bf16 and/or f32 output
// ---------------------------------------------------------------------------
__global__ void ln_kernel(const float* __restrict__ x,
                          const float* __restrict__ g, const float* __restrict__ b,
                          __hip_bfloat16* __restrict__ outB, float* __restrict__ outF,
                          int D) {
  int row = blockIdx.x;
  const float* xr = x + (size_t)row * D;
  float s = 0.f, s2 = 0.f;
  for (int i = threadIdx.x; i < D; i += blockDim.x) {
    float v = xr[i];
    s += v; s2 += v * v;
  }
  __shared__ float red[2][8];
#pragma unroll
  for (int off = 16; off > 0; off >>= 1) {
    s  += __shfl_down(s,  off, 32);
    s2 += __shfl_down(s2, off, 32);
  }
  int wid = threadIdx.x >> 5, lid = threadIdx.x & 31;
  if (lid == 0) { red[0][wid] = s; red[1][wid] = s2; }
  __syncthreads();
  if (threadIdx.x == 0) {
    float ts = 0.f, ts2 = 0.f;
    int nw = blockDim.x >> 5;
    for (int i = 0; i < nw; ++i) { ts += red[0][i]; ts2 += red[1][i]; }
    red[0][0] = ts; red[1][0] = ts2;
  }
  __syncthreads();
  float mean = red[0][0] / (float)D;
  float var  = red[1][0] / (float)D - mean * mean;
  float inv  = rsqrtf(var + 1e-5f);
  for (int i = threadIdx.x; i < D; i += blockDim.x) {
    float v = (xr[i] - mean) * inv * g[i] + b[i];
    if (outB) outB[(size_t)row * D + i] = __float2bfloat16(v);
    if (outF) outF[(size_t)row * D + i] = v;
  }
}

// ---------------------------------------------------------------------------
// bf16 WMMA GEMM:  out = A[M,K] @ Bt[N,K]^T + bias (+residual)(relu)
// Block 256 threads = 8 waves; block tile 128x128; wave tile 32x64 (2x4 WMMA tiles).
// K-slices (128x32) of A and B are double-buffered in LDS via
// global_load_async_to_lds_b128; WMMA fragments come from ds_read_b128.
// ---------------------------------------------------------------------------
__global__ __launch_bounds__(256)
void gemm_bf16(const __hip_bfloat16* __restrict__ A, const __hip_bfloat16* __restrict__ Bt,
               const float* __restrict__ bias, const float* __restrict__ residual,
               float* __restrict__ outF, __hip_bfloat16* __restrict__ outB,
               int M, int N, int K, int relu) {
  __shared__ __align__(16) __hip_bfloat16 Asmem[2][128 * LDS_STRIDE];
  __shared__ __align__(16) __hip_bfloat16 Bsmem[2][128 * LDS_STRIDE];

  const int tid  = threadIdx.x;
  const int lane = tid & 31;
  const int wave = tid >> 5;
  const int wm = wave & 3;            // 4 waves along M
  const int wn = wave >> 2;           // 2 waves along N
  const int rc = lane & 15;           // fragment row (A) / column (B,D)
  const int hi = lane >> 4;

  const __hip_bfloat16* Ab = A  + (size_t)(blockIdx.y * 128) * K;  // 128 rows of A
  const __hip_bfloat16* Bb = Bt + (size_t)(blockIdx.x * 128) * K;  // 128 cols of B (rows of Bt)

  // Per-thread async staging map: row = tid/2, two 16B chunks at (tid&1)*32 bytes.
  const int srow = tid >> 1;
  const int scol = (tid & 1) * 16;            // element offset within 32-wide K slice
  const uint64_t gA = (uint64_t)(const void*)(Ab + (size_t)srow * K + scol);
  const uint64_t gB = (uint64_t)(const void*)(Bb + (size_t)srow * K + scol);
  const uint32_t lOffA0 = (uint32_t)(size_t)(const void*)(&Asmem[0][srow * LDS_STRIDE + scol]);
  const uint32_t lOffA1 = (uint32_t)(size_t)(const void*)(&Asmem[1][srow * LDS_STRIDE + scol]);
  const uint32_t lOffB0 = (uint32_t)(size_t)(const void*)(&Bsmem[0][srow * LDS_STRIDE + scol]);
  const uint32_t lOffB1 = (uint32_t)(size_t)(const void*)(&Bsmem[1][srow * LDS_STRIDE + scol]);

  auto stage = [&](int k, int buf) {
    uint64_t ga = gA + (size_t)k * 2;         // bf16 -> bytes
    uint64_t gb = gB + (size_t)k * 2;
    uint32_t la = buf ? lOffA1 : lOffA0;
    uint32_t lb = buf ? lOffB1 : lOffB0;
    async_load_b128(ga,      la);
    async_load_b128(ga + 16, la + 16);
    async_load_b128(gb,      lb);
    async_load_b128(gb + 16, lb + 16);
  };

  v8f acc[2][4];
#pragma unroll
  for (int i = 0; i < 2; ++i)
#pragma unroll
    for (int j = 0; j < 4; ++j)
      acc[i][j] = (v8f){0.f, 0.f, 0.f, 0.f, 0.f, 0.f, 0.f, 0.f};

  const int nsteps = K >> 5;
  stage(0, 0);

  for (int s = 0; s < nsteps; ++s) {
    wait_async0();          // my async writes done
    __syncthreads();        // everyone's writes done; everyone's prior reads done
    if (s + 1 < nsteps) stage((s + 1) << 5, (s + 1) & 1);

    const __hip_bfloat16* As = Asmem[s & 1];
    const __hip_bfloat16* Bs = Bsmem[s & 1];
    bfrag a[2], bb[4];
#pragma unroll
    for (int i = 0; i < 2; ++i) {
      const __hip_bfloat16* p = As + (wm * 32 + i * 16 + rc) * LDS_STRIDE;
      a[i] = load_frag(p + hi * 8, p + 16 + hi * 8);
    }
#pragma unroll
    for (int j = 0; j < 4; ++j) {
      const __hip_bfloat16* p = Bs + (wn * 64 + j * 16 + rc) * LDS_STRIDE;
      bb[j] = load_frag(p + hi * 8, p + 16 + hi * 8);
    }
#pragma unroll
    for (int i = 0; i < 2; ++i)
#pragma unroll
      for (int j = 0; j < 4; ++j)
        acc[i][j] = wmma_bf16(a[i], bb[j], acc[i][j]);
  }

  // Epilogue: D tile layout — lane&15 = column, VGPR r = row r + 8*(lane>>4)
  const int m0 = blockIdx.y * 128 + wm * 32;
  const int n0 = blockIdx.x * 128 + wn * 64;
#pragma unroll
  for (int i = 0; i < 2; ++i) {
#pragma unroll
    for (int j = 0; j < 4; ++j) {
      int col = n0 + j * 16 + rc;
      float bv = bias ? bias[col] : 0.f;
#pragma unroll
      for (int r = 0; r < 8; ++r) {
        int mrow = m0 + i * 16 + hi * 8 + r;
        float v = acc[i][j][r] + bv;
        if (relu) v = fmaxf(v, 0.f);
        size_t idx = (size_t)mrow * N + col;
        if (residual) v += residual[idx];
        if (outF) outF[idx] = v;
        if (outB) outB[idx] = __float2bfloat16(v);
      }
    }
  }
}

// ---------------------------------------------------------------------------
// Chunked local attention: one wave per (chunk, head, batch).
// 16 queries x up to 80 keys x DK=64. scores = qu.k^T + qv.p^T (WMMA),
// softmax in LDS, out = attn @ v (WMMA, V transposed into LDS, zero-padded).
// ---------------------------------------------------------------------------
__global__ __launch_bounds__(32)
void attn_kernel(const __hip_bfloat16* __restrict__ qu, const __hip_bfloat16* __restrict__ qv,
                 const __hip_bfloat16* __restrict__ kk, const __hip_bfloat16* __restrict__ vv,
                 const __hip_bfloat16* __restrict__ pp, __hip_bfloat16* __restrict__ oo) {
  const int ci = blockIdx.x, h = blockIdx.y, b = blockIdx.z;
  const int lane = threadIdx.x;
  const int rc = lane & 15;
  const int hi = lane >> 4;
  const int nkc = (ci < 4 ? ci : 4) + 1;          // valid key chunks
  const int s0 = (ci + 1 - nkc) * CS_;            // window start
  const int q0 = ci * CS_;
  const int nk = nkc * CS_;                       // valid keys (16..80)
  const int nkp = (nk + 31) & ~31;                // padded to K multiple of 32
  const float scale = 0.125f;                     // 1/sqrt(DK)

  __shared__ __align__(16) float          sc[16][80];
  __shared__ __align__(16) __hip_bfloat16 pr[16][96];
  __shared__ __align__(16) __hip_bfloat16 vT[64][96];

  auto idx4 = [&](int t, int d) -> size_t {
    return (((size_t)b * T_ + t) * H_ + h) * DK_ + d;
  };

  const __hip_bfloat16* quR = qu + idx4(q0 + rc, 0);
  const __hip_bfloat16* qvR = qv + idx4(q0 + rc, 0);

  // ---- scores: (qu.k^T + qv.p^T) * scale, one 16x16 tile per key chunk ----
  for (int kc = 0; kc < nkc; ++kc) {
    int sbase = s0 + kc * CS_;
    const __hip_bfloat16* kR = kk + idx4(sbase + rc, 0);
    const __hip_bfloat16* pR = pp + (size_t)(sbase + rc) * (H_ * DK_) + h * DK_;
    v8f acc = (v8f){0.f, 0.f, 0.f, 0.f, 0.f, 0.f, 0.f, 0.f};
#pragma unroll
    for (int ks = 0; ks < DK_; ks += 32) {
      bfrag a  = load_frag(quR + ks + hi * 8, quR + ks + 16 + hi * 8);
      bfrag bb = load_frag(kR  + ks + hi * 8, kR + ks + 16 + hi * 8);
      acc = wmma_bf16(a, bb, acc);
    }
#pragma unroll
    for (int ks = 0; ks < DK_; ks += 32) {
      bfrag a  = load_frag(qvR + ks + hi * 8, qvR + ks + 16 + hi * 8);
      bfrag bb = load_frag(pR  + ks + hi * 8, pR + ks + 16 + hi * 8);
      acc = wmma_bf16(a, bb, acc);
    }
#pragma unroll
    for (int r = 0; r < 8; ++r)
      sc[r + hi * 8][kc * 16 + rc] = acc[r] * scale;
  }
  __syncthreads();

  // ---- softmax over the window (whole chunks valid; rest is -inf) ----
  if (lane < 16) {
    float mx = -1e30f;
    for (int c = 0; c < nk; ++c) mx = fmaxf(mx, sc[lane][c]);
    float sum = 0.f;
    for (int c = 0; c < nk; ++c) { float e = __expf(sc[lane][c] - mx); sc[lane][c] = e; sum += e; }
    float inv = 1.f / sum;
    for (int c = 0; c < 96; ++c)
      pr[lane][c] = __float2bfloat16(c < nk ? sc[lane][c] * inv : 0.f);
  }
  // ---- stage V transposed (zero-pad to nkp keys) ----
  for (int t = lane; t < nkp * DK_; t += 32) {
    int s = t >> 6, d = t & 63;
    vT[d][s] = (s < nk) ? vv[idx4(s0 + s, d)] : __float2bfloat16(0.f);
  }
  __syncthreads();

  // ---- out = attn @ v : 16 x 64 output, K = nkp ----
#pragma unroll
  for (int nt = 0; nt < 4; ++nt) {
    v8f acc = (v8f){0.f, 0.f, 0.f, 0.f, 0.f, 0.f, 0.f, 0.f};
    const __hip_bfloat16* prow = &pr[rc][0];
    const __hip_bfloat16* vrow = &vT[nt * 16 + rc][0];
    for (int ks = 0; ks < nkp; ks += 32) {
      bfrag a  = load_frag(prow + ks + hi * 8, prow + ks + 16 + hi * 8);
      bfrag bb = load_frag(vrow + ks + hi * 8, vrow + ks + 16 + hi * 8);
      acc = wmma_bf16(a, bb, acc);
    }
#pragma unroll
    for (int r = 0; r < 8; ++r)
      oo[idx4(q0 + hi * 8 + r, nt * 16 + rc)] = __float2bfloat16(acc[r]);
  }
}

// ---------------------------------------------------------------------------
// Host orchestration
// ---------------------------------------------------------------------------
extern "C" void kernel_launch(void* const* d_in, const int* in_sizes, int n_in,
                              void* d_out, int out_size, void* d_ws, size_t ws_size,
                              hipStream_t stream) {
  (void)in_sizes; (void)n_in; (void)out_size; (void)ws_size;
  const float* xs      = (const float*)d_in[0];
  const float* pos_emb = (const float*)d_in[1];
  /* d_in[2] = mask (bool) — structure is known, unused */
  const float* Wq = (const float*)d_in[3],  *bq = (const float*)d_in[4];
  const float* Wk = (const float*)d_in[5],  *bk = (const float*)d_in[6];
  const float* Wv = (const float*)d_in[7],  *bv = (const float*)d_in[8];
  const float* Wo = (const float*)d_in[9],  *bo = (const float*)d_in[10];
  const float* Wp = (const float*)d_in[11];
  const float* pbu = (const float*)d_in[12], *pbv = (const float*)d_in[13];
  const float* ln1_s = (const float*)d_in[14], *ln1_b = (const float*)d_in[15];
  const float* ln2_s = (const float*)d_in[16], *ln2_b = (const float*)d_in[17];
  const float* W1 = (const float*)d_in[18], *b1 = (const float*)d_in[19];
  const float* W2 = (const float*)d_in[20], *b2 = (const float*)d_in[21];
  const float* lnf_s = (const float*)d_in[22], *lnf_b = (const float*)d_in[23];

  char* ws = (char*)d_ws;
  size_t off = 0;
  auto take = [&](size_t bytes) -> void* {
    void* p = ws + off;
    off += (bytes + 255) & ~(size_t)255;
    return p;
  };
  const size_t BT = (size_t)B_ * T_;

  float*          x     = (float*)take(BT * D_ * 4);
  __hip_bfloat16* hbf   = (__hip_bfloat16*)take(BT * D_ * 2);
  float*          qf    = (float*)take(BT * D_ * 4);
  __hip_bfloat16* qubf  = (__hip_bfloat16*)take(BT * D_ * 2);
  __hip_bfloat16* qvbf  = (__hip_bfloat16*)take(BT * D_ * 2);
  __hip_bfloat16* kbf   = (__hip_bfloat16*)take(BT * D_ * 2);
  __hip_bfloat16* vbf   = (__hip_bfloat16*)take(BT * D_ * 2);
  __hip_bfloat16* pbf   = (__hip_bfloat16*)take((size_t)T_ * D_ * 2);
  __hip_bfloat16* obf   = (__hip_bfloat16*)take(BT * D_ * 2);
  __hip_bfloat16* ubf   = (__hip_bfloat16*)take(BT * FF_ * 2);
  __hip_bfloat16* posbf = (__hip_bfloat16*)take((size_t)T_ * D_ * 2);
  __hip_bfloat16* WqT = (__hip_bfloat16*)take((size_t)L_ * D_ * D_ * 2);
  __hip_bfloat16* WkT = (__hip_bfloat16*)take((size_t)L_ * D_ * D_ * 2);
  __hip_bfloat16* WvT = (__hip_bfloat16*)take((size_t)L_ * D_ * D_ * 2);
  __hip_bfloat16* WoT = (__hip_bfloat16*)take((size_t)L_ * D_ * D_ * 2);
  __hip_bfloat16* WpT = (__hip_bfloat16*)take((size_t)L_ * D_ * D_ * 2);
  __hip_bfloat16* W1T = (__hip_bfloat16*)take((size_t)L_ * D_ * FF_ * 2);
  __hip_bfloat16* W2T = (__hip_bfloat16*)take((size_t)L_ * FF_ * D_ * 2);

  dim3 tb(32, 8);
  transpose_to_bf16<<<dim3(D_ / 32, D_ / 32, L_), tb, 0, stream>>>(Wq, WqT, D_, D_);
  transpose_to_bf16<<<dim3(D_ / 32, D_ / 32, L_), tb, 0, stream>>>(Wk, WkT, D_, D_);
  transpose_to_bf16<<<dim3(D_ / 32, D_ / 32, L_), tb, 0, stream>>>(Wv, WvT, D_, D_);
  transpose_to_bf16<<<dim3(D_ / 32, D_ / 32, L_), tb, 0, stream>>>(Wo, WoT, D_, D_);
  transpose_to_bf16<<<dim3(D_ / 32, D_ / 32, L_), tb, 0, stream>>>(Wp, WpT, D_, D_);
  transpose_to_bf16<<<dim3(FF_ / 32, D_ / 32, L_), tb, 0, stream>>>(W1, W1T, D_, FF_);
  transpose_to_bf16<<<dim3(D_ / 32, FF_ / 32, L_), tb, 0, stream>>>(W2, W2T, FF_, D_);

  const int n = (int)(BT * D_);
  scale_kernel<<<(n + 255) / 256, 256, 0, stream>>>(xs, x, n, sqrtf((float)D_));
  cvt_bf16_kernel<<<(T_ * D_ + 255) / 256, 256, 0, stream>>>(pos_emb, posbf, T_ * D_);

  auto gemm = [&](const __hip_bfloat16* A, const __hip_bfloat16* Bt, const float* bias,
                  const float* res, float* oF, __hip_bfloat16* oB,
                  int M, int N, int K, int relu) {
    gemm_bf16<<<dim3(N / 128, M / 128), 256, 0, stream>>>(A, Bt, bias, res, oF, oB, M, N, K, relu);
  };

  for (int l = 0; l < L_; ++l) {
    const size_t wdd = (size_t)l * D_ * D_;
    ln_kernel<<<(int)BT, 256, 0, stream>>>(x, ln1_s + l * D_, ln1_b + l * D_, hbf, nullptr, D_);
    gemm(hbf,   WqT + wdd, bq + l * D_, nullptr, qf,      nullptr, (int)BT, D_, D_, 0);
    gemm(hbf,   WkT + wdd, bk + l * D_, nullptr, nullptr, kbf,     (int)BT, D_, D_, 0);
    gemm(hbf,   WvT + wdd, bv + l * D_, nullptr, nullptr, vbf,     (int)BT, D_, D_, 0);
    gemm(posbf, WpT + wdd, nullptr,     nullptr, nullptr, pbf,     T_,      D_, D_, 0);
    quqv_kernel<<<(n + 255) / 256, 256, 0, stream>>>(qf, pbu + l * D_, pbv + l * D_,
                                                     qubf, qvbf, n, D_);
    attn_kernel<<<dim3(T_ / CS_, H_, B_), 32, 0, stream>>>(qubf, qvbf, kbf, vbf, pbf, obf);
    gemm(obf, WoT + wdd, bo + l * D_, x, x, nullptr, (int)BT, D_, D_, 0);   // x += o@Wo + bo
    ln_kernel<<<(int)BT, 256, 0, stream>>>(x, ln2_s + l * D_, ln2_b + l * D_, hbf, nullptr, D_);
    gemm(hbf, W1T + (size_t)l * D_ * FF_, b1 + l * FF_, nullptr, nullptr, ubf,
         (int)BT, FF_, D_, 1);                                              // relu(h2@W1+b1)
    gemm(ubf, W2T + (size_t)l * FF_ * D_, b2 + l * D_, x, x, nullptr,
         (int)BT, D_, FF_, 0);                                              // x += u@W2 + b2
  }
  ln_kernel<<<(int)BT, 256, 0, stream>>>(x, lnf_s, lnf_b, nullptr, (float*)d_out, D_);
}
